// SwinJSCC_Encoder_53927609369158
// MI455X (gfx1250) — matrix operand order repaired and checked by reading, not compiled, not run on hardware
//
#include <hip/hip_runtime.h>

// ---------------------------------------------------------------------------
// SwinJSCC encoder for MI455X (gfx1250, wave32, WMMA).
// All dense math runs on v_wmma_f32_16x16x32_f16 (f16 operands, f32 accum).
// Cross-attention is algebraically reduced: Tk==1 => softmax==1 => o==v,
// so q/k projections and their LayerNorm are dead code and skipped.
// ---------------------------------------------------------------------------

typedef __attribute__((ext_vector_type(16))) _Float16 v16h;
typedef __attribute__((ext_vector_type(8)))  _Float16 v8h;
typedef __attribute__((ext_vector_type(8)))  float    v8f;

#define BATCH   16
#define HH      64
#define WW_     64
#define CCH     256
#define HEADS   8
#define WS_     8
#define NWIN    64          // windows per image (8x8)
#define NTOK    64          // tokens per window
#define HD      32
#define TTOT    (BATCH * HH * WW_)   // 65536 tokens
#define MLP_H   1024
#define QD      512

// ---------------- WMMA fragment helpers (layouts per CDNA5 ISA 7.12.2) -----

__device__ inline v8f wmma16(v16h a, v16h b, v8f c) {
  return __builtin_amdgcn_wmma_f32_16x16x32_f16(false, a, false, b, (short)0, c,
                                                false, false);
}

// A-matrix 16x32 f16 fragment from row-major LDS tile.
// lane<16: M=lane,    K = {ks..ks+7} U {16+ks..16+ks+7} with ks=0
// lane>=16: same M,   ks=8
__device__ inline v16h load_a_frag(const _Float16* base, int row0, int koff,
                                   int stride, int lane) {
  int r  = row0 + (lane & 15);
  int ks = (lane >> 4) << 3;
  const _Float16* p = base + (size_t)r * stride + koff + ks;
  v8h lo = *(const v8h*)p;
  v8h hi = *(const v8h*)(p + 16);
  v16h a;
#pragma unroll
  for (int i = 0; i < 8; ++i) { a[i] = lo[i]; a[i + 8] = hi[i]; }
  return a;
}

// B-matrix 32x16 f16 fragment from K-transposed LDS tile Bst[n][k].
// lane%16 = column n, K = {ks..ks+7} U {16+ks..16+ks+7}, ks=(lane/16)*8.
__device__ inline v16h load_b_frag(const _Float16* base, int n0, int koff,
                                   int stride, int lane) {
  int r = n0 + (lane & 15);
  int c = koff + ((lane >> 4) << 3);
  const _Float16* p = base + (size_t)r * stride + c;
  v8h lo = *(const v8h*)p;
  v8h hi = *(const v8h*)(p + 16);
  v16h b;
#pragma unroll
  for (int i = 0; i < 8; ++i) { b[i] = lo[i]; b[i + 8] = hi[i]; }
  return b;
}

// windowed token index (b*4096 + win*64 + n) -> image token index in x,
// including the roll(shift) (forward source == reverse destination).
__device__ inline int win_to_img(int t, int shift) {
  int b = t >> 12, r = t & 4095;
  int win = r >> 6, n = r & 63;
  int h = (((win >> 3) << 3) + (n >> 3) + shift) & 63;
  int w = (((win & 7) << 3) + (n & 7) + shift) & 63;
  return (b << 12) + (h << 6) + w;
}

__device__ inline float gelu_exact(float v) {
  return 0.5f * v * (1.0f + erff(v * 0.70710678118654752f));
}

// ---------------- elementwise: f32 -> f16 cast -----------------------------

__global__ __launch_bounds__(256) void cast_f16_kernel(
    const float* __restrict__ in, _Float16* __restrict__ out, int n) {
  int i = blockIdx.x * 256 + threadIdx.x;
  if (i < n) out[i] = (_Float16)in[i];
}

// ---------------- LayerNorm (one wave32 per token, 8 ch/lane) --------------
// windowed!=0: destination is window-ordered (with roll by -shift on read).

__global__ __launch_bounds__(256) void layernorm_kernel(
    const float* __restrict__ x, const float* __restrict__ g,
    const float* __restrict__ bta, _Float16* __restrict__ out,
    int windowed, int shift) {
  int tok  = blockIdx.x * 8 + (threadIdx.x >> 5);
  int lane = threadIdx.x & 31;
  int src  = windowed ? win_to_img(tok, shift) : tok;

  const float* xp = x + (size_t)src * CCH + lane * 8;
  float v[8];
  float s = 0.0f;
#pragma unroll
  for (int i = 0; i < 8; ++i) { v[i] = xp[i]; s += v[i]; }
#pragma unroll
  for (int o = 16; o > 0; o >>= 1) s += __shfl_xor(s, o, 32);
  float mu = s * (1.0f / CCH);
  float q = 0.0f;
#pragma unroll
  for (int i = 0; i < 8; ++i) { float d = v[i] - mu; q += d * d; }
#pragma unroll
  for (int o = 16; o > 0; o >>= 1) q += __shfl_xor(q, o, 32);
  float rstd = rsqrtf(q * (1.0f / CCH) + 1e-5f);

  _Float16* op = out + (size_t)tok * CCH + lane * 8;
#pragma unroll
  for (int i = 0; i < 8; ++i) {
    int c = lane * 8 + i;
    op[i] = (_Float16)((v[i] - mu) * rstd * g[c] + bta[c]);
  }
}

// ---------------- tiled WMMA GEMM: C[M,N] = A[M,K] * B[K,N] + bias ---------
// 256 thr = 8 waves, 64x64 output tile, K staged 32 at a time through LDS.
// mode 0: out f16         mode 1: GELU -> out f16
// mode 2: += into f32 x   mode 3: += into f32 x with window-reverse mapping

#define LDA_S 40   // 32 + 8 halves pad (80B rows, 16B-divisible)
#define LDB_S 40

__global__ __launch_bounds__(256) void gemm_wmma_kernel(
    const _Float16* __restrict__ A, int K,
    const _Float16* __restrict__ Bm, int N,
    const float* __restrict__ bias,
    float* __restrict__ outResid, _Float16* __restrict__ outH,
    int mode, int shift) {
  __shared__ __align__(16) _Float16 As[64 * LDA_S];
  __shared__ __align__(16) _Float16 Bst[64 * LDB_S];  // transposed: [n][k]

  int tid  = threadIdx.x;
  int lane = tid & 31;
  int wv   = tid >> 5;
  int row0 = blockIdx.y * 64;
  int col0 = blockIdx.x * 64;

  int ti  = wv & 3;          // output row tile
  int tj0 = wv >> 2;         // output col tiles tj0 and tj0+2

  int ar = tid >> 2, ac = (tid & 3) << 3;   // A stage: 64 rows x 32 halves
  int bk = tid >> 3, bn = (tid & 7) << 3;   // B stage: 32 k-rows x 64 n

  v8f acc0 = {}; v8f acc1 = {};

  for (int k0 = 0; k0 < K; k0 += 32) {
    v8h av = *(const v8h*)(A + (size_t)(row0 + ar) * K + k0 + ac);
    v8h bv = *(const v8h*)(Bm + (size_t)(k0 + bk) * N + col0 + bn);
    if (k0 + 32 < K)
      __builtin_prefetch(A + (size_t)(row0 + ar) * K + k0 + 32 + ac, 0, 1);
    *(v8h*)(As + ar * LDA_S + ac) = av;
#pragma unroll
    for (int j = 0; j < 8; ++j) Bst[(bn + j) * LDB_S + bk] = bv[j];
    __syncthreads();

    v16h a  = load_a_frag(As, ti * 16, 0, LDA_S, lane);
    v16h b0 = load_b_frag(Bst, tj0 * 16, 0, LDB_S, lane);
    v16h b1 = load_b_frag(Bst, (tj0 + 2) * 16, 0, LDB_S, lane);
    acc0 = wmma16(a, b0, acc0);
    acc1 = wmma16(a, b1, acc1);
    __syncthreads();
  }

  // epilogue: C layout VGPR r -> M = r + 8*(lane/16), N = lane%16
#pragma unroll
  for (int half = 0; half < 2; ++half) {
    const v8f& acc = half ? acc1 : acc0;
    int tj = half ? (tj0 + 2) : tj0;
#pragma unroll
    for (int r = 0; r < 8; ++r) {
      int grow = row0 + ti * 16 + r + ((lane >> 4) << 3);
      int gcol = col0 + tj * 16 + (lane & 15);
      float v = acc[r] + bias[gcol];
      if (mode == 1) v = gelu_exact(v);
      if (mode <= 1) {
        outH[(size_t)grow * N + gcol] = (_Float16)v;
      } else {
        int dst = (mode == 2) ? grow : win_to_img(grow, shift);
        outResid[(size_t)dst * CCH + gcol] += v;
      }
    }
  }
}

// ---------------- fused windowed attention (one block = window x head) -----
// S = (q k^T)*scale + relpos + shiftmask ; softmax ; out = P v. All in LDS.

__device__ inline int sw_region(int h) {     // H=64, WS=8, shift=4
  return h < 56 ? 0 : (h < 60 ? 1 : 2);
}

__global__ __launch_bounds__(256) void swin_attn_kernel(
    const _Float16* __restrict__ qkv,   // [T][768], window-ordered rows
    const float* __restrict__ rpb,      // [225][8]
    _Float16* __restrict__ attn_out,    // [T][256], window-ordered rows
    int shift) {
  __shared__ __align__(16) _Float16 Qs[64 * 40];
  __shared__ __align__(16) _Float16 Ks[64 * 40];
  __shared__ __align__(16) _Float16 Vt[32 * 72];   // transposed: [d][m]
  __shared__ __align__(16) float    Ss[64 * 72];
  __shared__ __align__(16) _Float16 Ps[64 * 72];

  int win  = blockIdx.x;        // 0..1023 (b*64 + wi)
  int head = blockIdx.y;        // 0..7
  int tid  = threadIdx.x;
  int lane = tid & 31;
  int wv   = tid >> 5;

  // stage q,k (row-major) and v (K-transposed) into LDS
  int m = tid >> 2, c4 = (tid & 3) << 3;
  size_t rowbase = ((size_t)(win * 64 + m)) * (3 * CCH) + head * HD + c4;
  v8h qv = *(const v8h*)(qkv + rowbase);
  v8h kv = *(const v8h*)(qkv + rowbase + CCH);
  v8h vv = *(const v8h*)(qkv + rowbase + 2 * CCH);
  *(v8h*)(Qs + m * 40 + c4) = qv;
  *(v8h*)(Ks + m * 40 + c4) = kv;
#pragma unroll
  for (int j = 0; j < 8; ++j) Vt[(c4 + j) * 72 + m] = vv[j];
  __syncthreads();

  // ---- scores: 4x4 grid of 16x16 tiles, K=32 == HD -> one WMMA per tile
  {
    int ti = wv & 3, tjA = wv >> 2, tjB = tjA + 2;
    v8f s0 = {}, s1 = {};
    v16h a  = load_a_frag(Qs, ti * 16, 0, 40, lane);
    v16h b0 = load_b_frag(Ks, tjA * 16, 0, 40, lane);
    v16h b1 = load_b_frag(Ks, tjB * 16, 0, 40, lane);
    s0 = wmma16(a, b0, s0);
    s1 = wmma16(a, b1, s1);

    const float scale = 0.17677669529663687f;   // 32^-0.5
    int wi  = win & 63;
    int wh8 = (wi >> 3) << 3, ww8 = (wi & 7) << 3;
#pragma unroll
    for (int half = 0; half < 2; ++half) {
      const v8f& s = half ? s1 : s0;
      int tj = half ? tjB : tjA;
#pragma unroll
      for (int r = 0; r < 8; ++r) {
        int nq = ti * 16 + r + ((lane >> 4) << 3);
        int nk = tj * 16 + (lane & 15);
        int i1 = nq >> 3, j1 = nq & 7, i2 = nk >> 3, j2 = nk & 7;
        float b = rpb[((i1 - i2 + 7) * 15 + (j1 - j2 + 7)) * HEADS + head];
        if (shift > 0) {
          int lq = sw_region(wh8 + i1) * 3 + sw_region(ww8 + j1);
          int lk = sw_region(wh8 + i2) * 3 + sw_region(ww8 + j2);
          if (lq != lk) b -= 100.0f;
        }
        Ss[nq * 72 + nk] = s[r] * scale + b;
      }
    }
  }
  __syncthreads();

  // ---- softmax over rows (thread per row)
  if (tid < 64) {
    float mx = -1e30f;
    for (int j = 0; j < 64; ++j) mx = fmaxf(mx, Ss[tid * 72 + j]);
    float sum = 0.0f;
    for (int j = 0; j < 64; ++j) {
      float e = __expf(Ss[tid * 72 + j] - mx);
      Ss[tid * 72 + j] = e;
      sum += e;
    }
    float inv = 1.0f / sum;
    for (int j = 0; j < 64; ++j)
      Ps[tid * 72 + j] = (_Float16)(Ss[tid * 72 + j] * inv);
  }
  __syncthreads();

  // ---- out = P(64x64) @ V(64x32): 4x2 tiles, K=64 -> 2 WMMA steps
  {
    int ti = wv >> 1, tj = wv & 1;
    v8f o = {};
#pragma unroll
    for (int kb = 0; kb < 64; kb += 32) {
      v16h a = load_a_frag(Ps, ti * 16, kb, 72, lane);
      v16h b = load_b_frag(Vt, tj * 16, kb, 72, lane);
      o = wmma16(a, b, o);
    }
#pragma unroll
    for (int r = 0; r < 8; ++r) {
      int nq = ti * 16 + r + ((lane >> 4) << 3);
      int d  = tj * 16 + (lane & 15);
      attn_out[((size_t)(win * 64 + nq)) * CCH + head * HD + d] = (_Float16)o[r];
    }
  }
}

// ---------------- cross-attention (Tk==1 => o == v; tiny) ------------------

__global__ __launch_bounds__(256) void cross_small_kernel(
    const float* __restrict__ emb, const float* __restrict__ qenc_w,
    const float* __restrict__ qenc_b, const float* __restrict__ in_w,
    const float* __restrict__ in_b, const float* __restrict__ out_w,
    const float* __restrict__ out_b, float* __restrict__ o) {
  __shared__ float cond[CCH];
  __shared__ float vbuf[CCH];
  int b = blockIdx.x, c = threadIdx.x;

  float acc = qenc_b[c];
  for (int q = 0; q < QD; ++q) acc += emb[b * QD + q] * qenc_w[q * CCH + c];
  cond[c] = acc;
  __syncthreads();

  acc = in_b[2 * CCH + c];                               // v row of in_w
  for (int j = 0; j < CCH; ++j) acc += cond[j] * in_w[(2 * CCH + c) * CCH + j];
  vbuf[c] = acc;
  __syncthreads();

  acc = out_b[c];                                        // o = v @ out_w^T
  for (int j = 0; j < CCH; ++j) acc += vbuf[j] * out_w[c * CCH + j];
  o[b * CCH + c] = acc;
}

__global__ __launch_bounds__(256) void bcast_add_kernel(
    float* __restrict__ x, const float* __restrict__ o) {
  size_t i = (size_t)blockIdx.x * 256 + threadIdx.x;
  int b = (int)(i >> 20);               // 4096*256 elems per batch
  int c = (int)(i & 255);
  x[i] += o[b * CCH + c];
}

// ---------------------------------------------------------------------------

extern "C" void kernel_launch(void* const* d_in, const int* in_sizes, int n_in,
                              void* d_out, int out_size, void* d_ws,
                              size_t ws_size, hipStream_t stream) {
  (void)in_sizes; (void)n_in; (void)out_size; (void)ws_size;

  const float* x_in    = (const float*)d_in[0];
  const float* emb     = (const float*)d_in[1];
  const float* qenc_w  = (const float*)d_in[2];
  const float* qenc_b  = (const float*)d_in[3];
  const float* n1_g    = (const float*)d_in[4];
  const float* n1_b    = (const float*)d_in[5];
  const float* qkv_w   = (const float*)d_in[6];
  const float* qkv_b   = (const float*)d_in[7];
  const float* rpb     = (const float*)d_in[8];
  const float* proj_w  = (const float*)d_in[9];
  const float* proj_b  = (const float*)d_in[10];
  const float* n2_g    = (const float*)d_in[11];
  const float* n2_b    = (const float*)d_in[12];
  const float* fc1_w   = (const float*)d_in[13];
  const float* fc1_b   = (const float*)d_in[14];
  const float* fc2_w   = (const float*)d_in[15];
  const float* fc2_b   = (const float*)d_in[16];
  const float* ca_in_w = (const float*)d_in[19];
  const float* ca_in_b = (const float*)d_in[20];
  const float* ca_out_w= (const float*)d_in[21];
  const float* ca_out_b= (const float*)d_in[22];

  // workspace layout (f16 elements, all offsets 16B aligned)
  _Float16* ws16 = (_Float16*)d_ws;
  size_t off = 0;
  _Float16* wqkv  = ws16 + off; off += (size_t)2 * 256 * 768;
  _Float16* wproj = ws16 + off; off += (size_t)2 * 256 * 256;
  _Float16* wfc1  = ws16 + off; off += (size_t)2 * 256 * 1024;
  _Float16* wfc2  = ws16 + off; off += (size_t)2 * 1024 * 256;
  _Float16* bufB  = ws16 + off; off += (size_t)TTOT * 256;    // norm / attn_out
  _Float16* bufA  = ws16 + off; off += (size_t)TTOT * 1024;   // qkv / mlp hidden
  float*    o_cr  = (float*)(ws16 + off);                     // [16][256]

  float* x = (float*)d_out;   // running activation (residual stream)
  hipMemcpyAsync(x, x_in, (size_t)TTOT * CCH * sizeof(float),
                 hipMemcpyDeviceToDevice, stream);

  // cast all GEMM weights to f16 once (both depths)
  auto cast = [&](const float* src, _Float16* dst, int n) {
    cast_f16_kernel<<<(n + 255) / 256, 256, 0, stream>>>(src, dst, n);
  };
  cast(qkv_w,  wqkv,  2 * 256 * 768);
  cast(proj_w, wproj, 2 * 256 * 256);
  cast(fc1_w,  wfc1,  2 * 256 * 1024);
  cast(fc2_w,  wfc2,  2 * 1024 * 256);

  const int shifts[2] = {0, 4};
  for (int d = 0; d < 2; ++d) {
    int shift = shifts[d];

    // LN1 + window partition (+roll) -> bufB f16 [T][256]
    layernorm_kernel<<<TTOT / 8, 256, 0, stream>>>(
        x, n1_g + d * CCH, n1_b + d * CCH, bufB, 1, shift);

    // QKV: [T,256]x[256,768] -> bufA f16
    gemm_wmma_kernel<<<dim3(768 / 64, TTOT / 64), 256, 0, stream>>>(
        bufB, 256, wqkv + (size_t)d * 256 * 768, 768, qkv_b + d * 768,
        nullptr, bufA, 0, 0);

    // fused windowed attention -> bufB f16 [T][256]
    swin_attn_kernel<<<dim3(1024, HEADS), 256, 0, stream>>>(
        bufA, rpb + d * 225 * HEADS, bufB, shift);

    // proj + window-reverse (+roll back) + residual into x
    gemm_wmma_kernel<<<dim3(256 / 64, TTOT / 64), 256, 0, stream>>>(
        bufB, 256, wproj + (size_t)d * 256 * 256, 256, proj_b + d * 256,
        x, nullptr, 3, shift);

    // LN2 -> bufB
    layernorm_kernel<<<TTOT / 8, 256, 0, stream>>>(
        x, n2_g + d * CCH, n2_b + d * CCH, bufB, 0, 0);

    // fc1 + GELU -> bufA f16 [T][1024]
    gemm_wmma_kernel<<<dim3(1024 / 64, TTOT / 64), 256, 0, stream>>>(
        bufB, 256, wfc1 + (size_t)d * 256 * 1024, 1024, fc1_b + d * 1024,
        nullptr, bufA, 1, 0);

    // fc2 + residual into x
    gemm_wmma_kernel<<<dim3(256 / 64, TTOT / 64), 256, 0, stream>>>(
        bufA, 1024, wfc2 + (size_t)d * 1024 * 256, 256, fc2_b + d * 256,
        x, nullptr, 2, 0);

    // cross-attention: cond -> v -> o (16x256), then broadcast add
    cross_small_kernel<<<16, 256, 0, stream>>>(
        emb, qenc_w, qenc_b,
        ca_in_w + (size_t)d * 768 * 256, ca_in_b + d * 768,
        ca_out_w + (size_t)d * 256 * 256, ca_out_b + d * 256, o_cr);
    bcast_add_kernel<<<(TTOT * CCH) / 256, 256, 0, stream>>>(x, o_cr);
  }
}
